// ChrEmbed_38397007626738
// MI455X (gfx1250) — matrix operand order repaired and checked by reading, not compiled, not run on hardware
//
#include <hip/hip_runtime.h>

#define NCHR  22
#define UNITS 256
#define BATCH 512
#define TOTF  94000

typedef __attribute__((ext_vector_type(8)))  __bf16 v8bf;
typedef __attribute__((ext_vector_type(16))) __bf16 v16bf;
typedef __attribute__((ext_vector_type(8)))  float  v8f;

__constant__ int d_SIZ[NCHR] = {8000, 7800, 6600, 6200, 6000, 5800, 5200, 5000,
                                4400, 4600, 4500, 4400, 3400, 3200, 3000, 3000,
                                2800, 2700, 2200, 2200, 1400, 1600};
__constant__ int d_OFF[NCHR] = {0, 8000, 15800, 22400, 28600, 34600, 40400, 45600,
                                50600, 55000, 59600, 64100, 68500, 71900, 75100,
                                78100, 81100, 83900, 86600, 88800, 91000, 92400};

struct ChrArgs {
    const float* W[NCHR];
    const float* b[NCHR];
};

union FragU {
    v16bf v;
    v8bf  h[2];
};

__global__ __launch_bounds__(256) void chr_embed_wmma(const float* __restrict__ x,
                                                      ChrArgs args,
                                                      float* __restrict__ out) {
    constexpr int MT = 128;   // M tile
    constexpr int NT = 64;    // N tile
    constexpr int KC = 32;    // K chunk
    constexpr int AP = 40;    // padded LDS row length in bf16 (80 B -> bank-conflict-free)

    // double-buffered staging tiles (60 KB total)
    __shared__ __align__(16) __bf16 sAhi[2 * MT * AP];
    __shared__ __align__(16) __bf16 sAlo[2 * MT * AP];
    __shared__ __align__(16) __bf16 sBhi[2 * NT * AP];   // stored transposed: [n][k]
    __shared__ __align__(16) __bf16 sBlo[2 * NT * AP];

    const int  c  = blockIdx.z;
    const int  m0 = blockIdx.x * MT;
    const int  n0 = blockIdx.y * NT;
    const int  ks = d_SIZ[c];
    const long off = d_OFF[c];
    const float* __restrict__ Wp = args.W[c];

    const int tid  = threadIdx.x;
    const int lane = tid & 31;
    const int wv   = tid >> 5;     // 0..7 (wave32)
    const int wm   = wv >> 1;      // 0..3: wave's M quadrant
    const int wn   = wv & 1;       // 0..1: wave's N quadrant
    const int lr   = lane & 15;
    const int lh   = lane >> 4;    // half-wave select

    // staging-thread coordinates (constant across chunks)
    const int aRow[4] = {(tid + 0)   >> 3, (tid + 256) >> 3,
                         (tid + 512) >> 3, (tid + 768) >> 3};
    const int aKq  = (tid & 7) << 2;                 // same for all 4 slices
    const int bKl[2] = {(tid + 0) >> 4, (tid + 256) >> 4};
    const int bNq  = (tid & 15) << 2;

    float4 aReg[4];
    float4 bReg[2];

    auto loadChunk = [&](int kc) {
        #pragma unroll
        for (int r = 0; r < 4; ++r) {
            int k = kc + aKq;
            aReg[r] = make_float4(0.f, 0.f, 0.f, 0.f);
            if (k < ks)
                aReg[r] = *(const float4*)(x + (long)(m0 + aRow[r]) * TOTF + off + k);
        }
        #pragma unroll
        for (int r = 0; r < 2; ++r) {
            int k = kc + bKl[r];
            bReg[r] = make_float4(0.f, 0.f, 0.f, 0.f);
            if (k < ks)
                bReg[r] = *(const float4*)(Wp + (long)k * UNITS + n0 + bNq);
        }
    };

    v8f acc[2][2] = {};

    loadChunk(0);  // prologue: chunk 0 into registers

    for (int kc = 0, p = 0; kc < ks; kc += KC, p ^= 1) {
        // ---- convert staged registers -> bf16 hi/lo LDS tiles (buffer p) ----
        __bf16* __restrict__ pAhi = sAhi + p * MT * AP;
        __bf16* __restrict__ pAlo = sAlo + p * MT * AP;
        __bf16* __restrict__ pBhi = sBhi + p * NT * AP;
        __bf16* __restrict__ pBlo = sBlo + p * NT * AP;

        #pragma unroll
        for (int r = 0; r < 4; ++r) {
            float vv[4] = {aReg[r].x, aReg[r].y, aReg[r].z, aReg[r].w};
            int base = aRow[r] * AP + aKq;
            #pragma unroll
            for (int j = 0; j < 4; ++j) {
                __bf16 h = (__bf16)vv[j];
                pAhi[base + j] = h;
                pAlo[base + j] = (__bf16)(vv[j] - (float)h);
            }
        }
        #pragma unroll
        for (int r = 0; r < 2; ++r) {
            float vv[4] = {bReg[r].x, bReg[r].y, bReg[r].z, bReg[r].w};
            #pragma unroll
            for (int j = 0; j < 4; ++j) {
                __bf16 h = (__bf16)vv[j];
                pBhi[(bNq + j) * AP + bKl[r]] = h;
                pBlo[(bNq + j) * AP + bKl[r]] = (__bf16)(vv[j] - (float)h);
            }
        }

        __syncthreads();   // single barrier per chunk (double buffer)

        // ---- start global loads for chunk i+1: overlaps the WMMA block below ----
        if (kc + KC < ks)
            loadChunk(kc + KC);

        // ---- prefetch chunk i+2 toward L2 ----
        if (kc + 2 * KC < ks) {
            if (tid < 128) {
                __builtin_prefetch(x + (long)(m0 + tid) * TOTF + off + kc + 2 * KC, 0, 3);
            } else {
                int k2 = (tid - 128) >> 1;
                int hf = tid & 1;
                if (kc + 2 * KC + k2 < ks)
                    __builtin_prefetch(Wp + (long)(kc + 2 * KC + k2) * UNITS + n0 + hf * 32, 0, 3);
            }
        }

        // ---- fragments + WMMA from buffer p ----
        FragU ah[2], al[2];
        #pragma unroll
        for (int mt = 0; mt < 2; ++mt) {
            const __bf16* pA = &pAhi[(wm * 32 + mt * 16 + lr) * AP + lh * 8];
            ah[mt].h[0] = *(const v8bf*)pA;          // K = kb .. kb+7
            ah[mt].h[1] = *(const v8bf*)(pA + 16);   // K = kb+16 .. kb+23
            const __bf16* qA = &pAlo[(wm * 32 + mt * 16 + lr) * AP + lh * 8];
            al[mt].h[0] = *(const v8bf*)qA;
            al[mt].h[1] = *(const v8bf*)(qA + 16);
        }
        #pragma unroll
        for (int nt = 0; nt < 2; ++nt) {
            FragU bh, bl;
            const __bf16* pB = &pBhi[(wn * 32 + nt * 16 + lr) * AP + lh * 8];
            bh.h[0] = *(const v8bf*)pB;              // K = kb .. kb+7
            bh.h[1] = *(const v8bf*)(pB + 16);       // K = kb+16 .. kb+23
            const __bf16* qB = &pBlo[(wn * 32 + nt * 16 + lr) * AP + lh * 8];
            bl.h[0] = *(const v8bf*)qB;
            bl.h[1] = *(const v8bf*)(qB + 16);
            #pragma unroll
            for (int mt = 0; mt < 2; ++mt) {
                // bf16x3 split-compensated fp32 product: ah*bh + ah*bl + al*bh
                acc[mt][nt] = __builtin_amdgcn_wmma_f32_16x16x32_bf16(
                    false, ah[mt].v, false, bh.v, (short)0, acc[mt][nt], false, false);
                acc[mt][nt] = __builtin_amdgcn_wmma_f32_16x16x32_bf16(
                    false, ah[mt].v, false, bl.v, (short)0, acc[mt][nt], false, false);
                acc[mt][nt] = __builtin_amdgcn_wmma_f32_16x16x32_bf16(
                    false, al[mt].v, false, bh.v, (short)0, acc[mt][nt], false, false);
            }
        }
    }

    // ---- epilogue: bias add + store; C/D layout: elem j -> M = j + 8*lh, N = lr ----
    #pragma unroll
    for (int nt = 0; nt < 2; ++nt) {
        int n = n0 + wn * 32 + nt * 16 + lr;
        float bias = args.b[c][n];
        #pragma unroll
        for (int mt = 0; mt < 2; ++mt) {
            int mBase = m0 + wm * 32 + mt * 16 + lh * 8;
            #pragma unroll
            for (int j = 0; j < 8; ++j) {
                out[((long)(mBase + j) * NCHR + c) * UNITS + n] = acc[mt][nt][j] + bias;
            }
        }
    }
}

extern "C" void kernel_launch(void* const* d_in, const int* in_sizes, int n_in,
                              void* d_out, int out_size, void* d_ws, size_t ws_size,
                              hipStream_t stream) {
    const float* x = (const float*)d_in[0];
    ChrArgs args;
    for (int i = 0; i < NCHR; ++i) {
        args.W[i] = (const float*)d_in[1 + i];
        args.b[i] = (const float*)d_in[1 + NCHR + i];
    }
    float* out = (float*)d_out;

    dim3 grid(BATCH / 128, UNITS / 64, NCHR);  // 4 x 4 x 22
    chr_embed_wmma<<<grid, dim3(256), 0, stream>>>(x, args, out);
}
